// GeneralConv_18940805775383
// MI455X (gfx1250) — compile-verified, matrix operand order checked by reading, not run on hardware
//
#include <hip/hip_runtime.h>
#include <math.h>

// ---------------- problem constants (match reference) ----------------
#define NN 40000
#define EE 640000
#define DD 128
#define TT 3
#define RR 4
#define HH 8
#define DK 16

typedef __attribute__((ext_vector_type(16))) __bf16 v16bf;
typedef __attribute__((ext_vector_type(8)))  float  v8f;

// ---------------- helpers ----------------
__device__ inline unsigned f2ord(float f) {
    unsigned u = __float_as_uint(f);
    return (u & 0x80000000u) ? ~u : (u | 0x80000000u);
}
__device__ inline float ord2f(unsigned u) {
    return __uint_as_float((u & 0x80000000u) ? (u ^ 0x80000000u) : ~u);
}

// Load 16 bf16 for one lane of an A/B fragment (16x16x32 bf16 WMMA).
// base must already include +kk and +8*half. Element pair j holds
// K = (j<4 ? 2j : 16+2(j-4)) (+8*half folded into base).
__device__ inline v16bf load_frag(const __bf16* base) {
    v16bf f;
#pragma unroll
    for (int j = 0; j < 8; ++j) {
        int koff = (j < 4) ? (2 * j) : (16 + 2 * (j - 4));
        f[2 * j]     = base[koff];
        f[2 * j + 1] = base[koff + 1];
    }
    return f;
}

// Same, but only K=0..15 are real (DK=16 relation transforms); K=16..31 zero.
// With +8*half folded into base, that means element pairs j<4 are real.
__device__ inline v16bf load_frag_k16(const __bf16* base) {
    v16bf f;
#pragma unroll
    for (int j = 0; j < 4; ++j) {
        f[2 * j]     = base[2 * j];
        f[2 * j + 1] = base[2 * j + 1];
    }
#pragma unroll
    for (int j = 4; j < 8; ++j) {
        f[2 * j]     = (__bf16)0.0f;
        f[2 * j + 1] = (__bf16)0.0f;
    }
    return f;
}

// ---------------- conversion kernels ----------------
__global__ __launch_bounds__(256) void k_cvt_bf16(const float* __restrict__ in,
                                                  __bf16* __restrict__ out, int n) {
    int i = blockIdx.x * 256 + threadIdx.x;
    if (i < n) out[i] = (__bf16)in[i];
}

// W[t][in][out] (f32) -> Wt[t][out][in] (bf16): contiguous-K B-fragment loads
__global__ __launch_bounds__(256) void k_cvtT_w(const float* __restrict__ W,
                                                __bf16* __restrict__ Wt) {
    int i = blockIdx.x * 256 + threadIdx.x;          // T*D*D threads
    int t = i / (DD * DD);
    int rem = i % (DD * DD);
    int ii = rem / DD, oo = rem % DD;
    Wt[(size_t)t * DD * DD + (size_t)oo * DD + ii] = (__bf16)W[i];
}

// rel[r][h][d][f] (f32) -> relT[r][h][f][d] (bf16)
__global__ __launch_bounds__(256) void k_cvtT_rel(const float* __restrict__ Rm,
                                                  __bf16* __restrict__ Rt) {
    int i = blockIdx.x * 256 + threadIdx.x;          // R*H*256 threads
    int rh = i / (DK * DK);
    int rem = i % (DK * DK);
    int d = rem / DK, f = rem % DK;
    Rt[(size_t)rh * DK * DK + f * DK + d] = (__bf16)Rm[i];
}

// ---------------- per-type projection GEMM (WMMA, bf16 -> f32 acc) ----------
// out[n,:] = X[n,:] @ W[type[n]] + b[type[n]] ; computed densely for all 3
// types per tile, then selected per row. grid = (N/16, D/16), block = 32.
__global__ __launch_bounds__(32)
void k_proj_wmma(const __bf16* __restrict__ X,    // [N,128] bf16
                 const __bf16* __restrict__ Wt,   // [T,128(out),128(in)] bf16
                 const float*  __restrict__ bias, // [T,128] f32
                 const int*    __restrict__ ntype,
                 float*  __restrict__ outF,       // [N,128] f32 or nullptr
                 __bf16* __restrict__ outB) {     // [N,128] bf16 or nullptr
    const int tileM = blockIdx.x;
    const int tileN = blockIdx.y;
    const int lane  = threadIdx.x;
    const int half  = lane >> 4;
    const int lm    = lane & 15;

    const int rowA = tileM * 16 + lm;      // A-matrix row held by this lane
    const int colg = tileN * 16 + lm;      // B/D column held by this lane

    v8f acc0 = {}, acc1 = {}, acc2 = {};
#pragma unroll
    for (int kk = 0; kk < DD; kk += 32) {
        const v16bf a = load_frag(X + (size_t)rowA * DD + kk + 8 * half);
        const __bf16* wb = Wt + (size_t)colg * DD + kk + 8 * half;
        v16bf b0 = load_frag(wb);
        v16bf b1 = load_frag(wb + (size_t)DD * DD);
        v16bf b2 = load_frag(wb + (size_t)2 * DD * DD);
        acc0 = __builtin_amdgcn_wmma_f32_16x16x32_bf16(false, a, false, b0, (short)0, acc0, false, false);
        acc1 = __builtin_amdgcn_wmma_f32_16x16x32_bf16(false, a, false, b1, (short)0, acc1, false, false);
        acc2 = __builtin_amdgcn_wmma_f32_16x16x32_bf16(false, a, false, b2, (short)0, acc2, false, false);
    }
#pragma unroll
    for (int rr = 0; rr < 8; ++rr) {
        const int grow = tileM * 16 + rr + 8 * half;   // D-matrix row
        const int t = ntype[grow];
        float v0 = acc0[rr], v1 = acc1[rr], v2 = acc2[rr];
        float val = (t == 0) ? v0 : ((t == 1) ? v1 : v2);
        val += bias[t * DD + colg];
        const size_t oi = (size_t)grow * DD + colg;
        if (outF) outF[oi] = val;
        if (outB) outB[oi] = (__bf16)val;
    }
}

// -------- per-(node, relation, head) DKxDK transforms (WMMA, K padded) ------
// krot[n,r,h*16+f] = sum_d k[n,h*16+d]*rel_att[r,h,d,f]; same for msg from v.
// grid = (N/16, H, R), block = 32.
__global__ __launch_bounds__(32)
void k_reltrans_wmma(const __bf16* __restrict__ Kbf,   // [N,128]
                     const __bf16* __restrict__ Vbf,   // [N,128]
                     const __bf16* __restrict__ AttT,  // [R,H,16(f),16(d)]
                     const __bf16* __restrict__ MsgT,  // [R,H,16(f),16(d)]
                     __bf16* __restrict__ krot,        // [N,R,128]
                     __bf16* __restrict__ msgall) {    // [N,R,128]
    const int tileM = blockIdx.x;
    const int h     = blockIdx.y;
    const int rel   = blockIdx.z;
    const int lane  = threadIdx.x;
    const int half  = lane >> 4;
    const int lm    = lane & 15;

    const int rowA = tileM * 16 + lm;
    const size_t nodeOff = (size_t)rowA * DD + h * DK + 8 * half;
    const v16bf ak = load_frag_k16(Kbf + nodeOff);
    const v16bf av = load_frag_k16(Vbf + nodeOff);

    const size_t relOff = (((size_t)rel * HH + h) * DK + lm) * DK + 8 * half;
    const v16bf ba = load_frag_k16(AttT + relOff);
    const v16bf bm = load_frag_k16(MsgT + relOff);

    v8f acck = {}, accm = {};
    acck = __builtin_amdgcn_wmma_f32_16x16x32_bf16(false, ak, false, ba, (short)0, acck, false, false);
    accm = __builtin_amdgcn_wmma_f32_16x16x32_bf16(false, av, false, bm, (short)0, accm, false, false);

#pragma unroll
    for (int rr = 0; rr < 8; ++rr) {
        const int grow = tileM * 16 + rr + 8 * half;
        const size_t oi = ((size_t)grow * RR + rel) * DD + h * DK + lm;
        krot[oi]   = (__bf16)acck[rr];
        msgall[oi] = (__bf16)accm[rr];
    }
}

// ---------------- edge pass A: attention logits + segment max --------------
__global__ __launch_bounds__(256)
void k_att(const int* __restrict__ src, const int* __restrict__ dst,
           const int* __restrict__ etype,
           const __bf16* __restrict__ Qbf,      // [N,128]
           const __bf16* __restrict__ krot,     // [N,R,128]
           const float* __restrict__ rel_pri,   // [R,H]
           float* __restrict__ att,             // [E,H]
           unsigned* __restrict__ segmax) {     // [N,H] (ordered-uint)
    const int idx = blockIdx.x * 256 + threadIdx.x;   // e*H + h
    if (idx >= EE * HH) return;
    const int e = idx >> 3, h = idx & 7;
    const int s = src[e], d = dst[e], rel = etype[e];
    const __bf16* qp = Qbf  + (size_t)d * DD + h * DK;
    const __bf16* kp = krot + ((size_t)s * RR + rel) * DD + h * DK;
    float sum = 0.f;
#pragma unroll
    for (int i = 0; i < DK; ++i) sum += (float)qp[i] * (float)kp[i];
    const float a = sum * rel_pri[rel * HH + h] * 0.25f;   // 1/sqrt(DK)
    att[idx] = a;
    atomicMax(&segmax[d * HH + h], f2ord(a));
}

// ---------------- edge pass B: exp + segment sum ---------------------------
__global__ __launch_bounds__(256)
void k_expsum(const int* __restrict__ src, const int* __restrict__ dst,
              const int* __restrict__ etype,
              const unsigned* __restrict__ segmax,
              const __bf16* __restrict__ msgall,
              float* __restrict__ att,           // in: logits, out: exp()
              float* __restrict__ denom) {       // [N,H]
    const int idx = blockIdx.x * 256 + threadIdx.x;
    if (idx >= EE * HH) return;
    const int e = idx >> 3, h = idx & 7;
    const int d = dst[e];
    const float m = ord2f(segmax[d * HH + h]);
    const float ex = expf(att[idx] - m);
    att[idx] = ex;
    atomicAdd(&denom[d * HH + h], ex);
    // warm L2 with the message row the aggregation pass will gather
    __builtin_prefetch(msgall + ((size_t)src[e] * RR + etype[e]) * DD + h * DK, 0, 1);
}

// ---------------- edge pass C: weighted message aggregation ----------------
__global__ __launch_bounds__(256)
void k_aggr(const int* __restrict__ src, const int* __restrict__ dst,
            const int* __restrict__ etype,
            const float* __restrict__ att,       // exp values [E,H]
            const float* __restrict__ denom,     // [N,H]
            const __bf16* __restrict__ msgall,   // [N,R,128]
            float* __restrict__ aggr) {          // [N,128]
    const long long idx = (long long)blockIdx.x * 256 + threadIdx.x;  // e*128+dd
    if (idx >= (long long)EE * DD) return;
    const int e  = (int)(idx >> 7);
    const int dd = (int)(idx & 127);
    const int h  = dd >> 4;
    const int s = src[e], d = dst[e], rel = etype[e];
    const float alpha = att[e * HH + h] / (denom[d * HH + h] + 1e-16f);
    const float val = (float)msgall[((size_t)s * RR + rel) * DD + dd] * alpha;
    atomicAdd(&aggr[(size_t)d * DD + dd], val);
}

// ---------------- update: exact GELU -> bf16 -------------------------------
__global__ __launch_bounds__(256)
void k_gelu(const float* __restrict__ aggr, __bf16* __restrict__ hbf, int n) {
    const int i = blockIdx.x * 256 + threadIdx.x;
    if (i >= n) return;
    const float x = aggr[i];
    const float g = 0.5f * x * (1.0f + erff(x * 0.70710678118654752f));
    hbf[i] = (__bf16)g;
}

// ---------------- skip-gate + per-type LayerNorm (1 wave per node) ---------
__global__ __launch_bounds__(256)
void k_layernorm(const float* __restrict__ trans,    // [N,128]
                 const float* __restrict__ node_inp, // [N,128]
                 const int*   __restrict__ ntype,
                 const float* __restrict__ skip,     // [T]
                 const float* __restrict__ ln_g,     // [T,128]
                 const float* __restrict__ ln_b,     // [T,128]
                 float* __restrict__ out) {
    const int wave = threadIdx.x >> 5;
    const int lane = threadIdx.x & 31;
    const int n = blockIdx.x * 8 + wave;
    if (n >= NN) return;
    const int t = ntype[n];
    const float a = 1.0f / (1.0f + expf(-skip[t]));

    const float4 tr = ((const float4*)(trans + (size_t)n * DD))[lane];
    const float4 xi = ((const float4*)(node_inp + (size_t)n * DD))[lane];
    float4 res;
    res.x = tr.x * a + xi.x * (1.0f - a);
    res.y = tr.y * a + xi.y * (1.0f - a);
    res.z = tr.z * a + xi.z * (1.0f - a);
    res.w = tr.w * a + xi.w * (1.0f - a);

    float s = res.x + res.y + res.z + res.w;
#pragma unroll
    for (int off = 16; off > 0; off >>= 1) s += __shfl_xor(s, off, 32);
    const float mu = s * (1.0f / 128.0f);

    const float dx = res.x - mu, dy = res.y - mu, dz = res.z - mu, dw = res.w - mu;
    float sq = dx * dx + dy * dy + dz * dz + dw * dw;
#pragma unroll
    for (int off = 16; off > 0; off >>= 1) sq += __shfl_xor(sq, off, 32);
    const float rins = rsqrtf(sq * (1.0f / 128.0f) + 1e-5f);

    const float4 g = ((const float4*)(ln_g + (size_t)t * DD))[lane];
    const float4 b = ((const float4*)(ln_b + (size_t)t * DD))[lane];
    float4 o;
    o.x = dx * rins * g.x + b.x;
    o.y = dy * rins * g.y + b.y;
    o.z = dz * rins * g.z + b.z;
    o.w = dw * rins * g.w + b.w;
    ((float4*)(out + (size_t)n * DD))[lane] = o;
}

// ---------------- launch ---------------------------------------------------
extern "C" void kernel_launch(void* const* d_in, const int* in_sizes, int n_in,
                              void* d_out, int out_size, void* d_ws, size_t ws_size,
                              hipStream_t stream) {
    const float* node_inp = (const float*)d_in[0];
    const int*   ntype    = (const int*)d_in[1];
    const int*   esrc     = (const int*)d_in[2];
    const int*   edst     = esrc + EE;
    const int*   etype    = (const int*)d_in[3];
    const float* Wk = (const float*)d_in[4];  const float* bk = (const float*)d_in[5];
    const float* Wq = (const float*)d_in[6];  const float* bq = (const float*)d_in[7];
    const float* Wv = (const float*)d_in[8];  const float* bv = (const float*)d_in[9];
    const float* Wa = (const float*)d_in[10]; const float* ba = (const float*)d_in[11];
    const float* rel_pri = (const float*)d_in[12];
    const float* rel_att = (const float*)d_in[13];
    const float* rel_msg = (const float*)d_in[14];
    const float* skip = (const float*)d_in[15];
    const float* ln_g = (const float*)d_in[16];
    const float* ln_b = (const float*)d_in[17];
    float* out = (float*)d_out;

    // workspace carve (256B aligned)
    char* w = (char*)d_ws;
    auto carve = [&](size_t bytes) -> void* {
        void* p = (void*)w;
        w += (bytes + 255) & ~(size_t)255;
        return p;
    };
    __bf16* xbf   = (__bf16*)carve((size_t)NN * DD * 2);
    __bf16* wtk   = (__bf16*)carve((size_t)TT * DD * DD * 2);
    __bf16* wtq   = (__bf16*)carve((size_t)TT * DD * DD * 2);
    __bf16* wtv   = (__bf16*)carve((size_t)TT * DD * DD * 2);
    __bf16* wta   = (__bf16*)carve((size_t)TT * DD * DD * 2);
    __bf16* rattT = (__bf16*)carve((size_t)RR * HH * DK * DK * 2);
    __bf16* rmsgT = (__bf16*)carve((size_t)RR * HH * DK * DK * 2);
    __bf16* kbf   = (__bf16*)carve((size_t)NN * DD * 2);
    __bf16* qbf   = (__bf16*)carve((size_t)NN * DD * 2);
    __bf16* vbf   = (__bf16*)carve((size_t)NN * DD * 2);
    __bf16* krot  = (__bf16*)carve((size_t)NN * RR * DD * 2);
    __bf16* msgal = (__bf16*)carve((size_t)NN * RR * DD * 2);
    float*  attb  = (float*)carve((size_t)EE * HH * 4);
    unsigned* smax = (unsigned*)carve((size_t)NN * HH * 4);
    float*  denom = (float*)carve((size_t)NN * HH * 4);
    float*  aggr  = (float*)carve((size_t)NN * DD * 4);
    __bf16* hbf   = (__bf16*)carve((size_t)NN * DD * 2);
    float*  trans = (float*)carve((size_t)NN * DD * 4);

    // precision conversions + weight transposes
    k_cvt_bf16<<<(NN * DD) / 256, 256, 0, stream>>>(node_inp, xbf, NN * DD);
    k_cvtT_w<<<(TT * DD * DD) / 256, 256, 0, stream>>>(Wk, wtk);
    k_cvtT_w<<<(TT * DD * DD) / 256, 256, 0, stream>>>(Wq, wtq);
    k_cvtT_w<<<(TT * DD * DD) / 256, 256, 0, stream>>>(Wv, wtv);
    k_cvtT_w<<<(TT * DD * DD) / 256, 256, 0, stream>>>(Wa, wta);
    k_cvtT_rel<<<(RR * HH * DK * DK) / 256, 256, 0, stream>>>(rel_att, rattT);
    k_cvtT_rel<<<(RR * HH * DK * DK) / 256, 256, 0, stream>>>(rel_msg, rmsgT);

    // per-type K/Q/V projections (WMMA)
    dim3 gproj(NN / 16, DD / 16);
    k_proj_wmma<<<gproj, 32, 0, stream>>>(xbf, wtk, bk, ntype, nullptr, kbf);
    k_proj_wmma<<<gproj, 32, 0, stream>>>(xbf, wtq, bq, ntype, nullptr, qbf);
    k_proj_wmma<<<gproj, 32, 0, stream>>>(xbf, wtv, bv, ntype, nullptr, vbf);

    // per-(node, relation) transformed keys/messages (WMMA)
    dim3 grel(NN / 16, HH, RR);
    k_reltrans_wmma<<<grel, 32, 0, stream>>>(kbf, vbf, rattT, rmsgT, krot, msgal);

    // segment-softmax stats must be re-zeroed every call (graph replays)
    hipMemsetAsync(smax, 0, (size_t)NN * HH * 4, stream);
    hipMemsetAsync(denom, 0, (size_t)NN * HH * 4, stream);
    hipMemsetAsync(aggr, 0, (size_t)NN * DD * 4, stream);

    k_att<<<(EE * HH) / 256, 256, 0, stream>>>(esrc, edst, etype, qbf, krot, rel_pri, attb, smax);
    k_expsum<<<(EE * HH) / 256, 256, 0, stream>>>(esrc, edst, etype, smax, msgal, attb, denom);
    k_aggr<<<(int)(((long long)EE * DD) / 256), 256, 0, stream>>>(esrc, edst, etype, attb, denom, msgal, aggr);

    // update path: gelu -> per-type A projection (WMMA) -> gate + LN
    k_gelu<<<(NN * DD) / 256, 256, 0, stream>>>(aggr, hbf, NN * DD);
    k_proj_wmma<<<gproj, 32, 0, stream>>>(hbf, wta, ba, ntype, trans, nullptr);
    k_layernorm<<<NN / 8, 256, 0, stream>>>(trans, node_inp, ntype, skip, ln_g, ln_b, out);
}
